// SelfAttention_91147795956156
// MI455X (gfx1250) — matrix operand order repaired and checked by reading, not compiled, not run on hardware
//
#include <hip/hip_runtime.h>

// ---------------------------------------------------------------------------
// Self-attention block (GroupNorm -> QKV -> flash attention -> proj + residual)
// for MI455X / gfx1250: wave32, WMMA f32_16x16x32_f16, async global->LDS copies.
// Shapes: B=4, C=256, N=64*64=4096, GROUPS=32.
// ---------------------------------------------------------------------------

#define B   4
#define C   256
#define C3  768
#define N   4096
#define GROUPS 32
#define EPS 1e-5f
#define SCALE 0.0625f   // 1/sqrt(256)

typedef __attribute__((ext_vector_type(16))) _Float16 v16h;
typedef __attribute__((ext_vector_type(8)))  _Float16 v8h;
typedef __attribute__((ext_vector_type(8)))  float    v8f;
typedef __attribute__((ext_vector_type(4)))  int      v4i;

// ---- CDNA5 async global->LDS path (ASYNCcnt) ------------------------------
#if __has_builtin(__builtin_amdgcn_global_load_async_to_lds_b128)
#define HAVE_ASYNC_LDS 1
typedef __attribute__((address_space(1))) v4i* gv4i_p;   // global int4*
typedef __attribute__((address_space(3))) v4i* lv4i_p;   // LDS int4*
__device__ __forceinline__ void async_b128(const _Float16* g, _Float16* l) {
  __builtin_amdgcn_global_load_async_to_lds_b128(
      (gv4i_p)(g), (lv4i_p)(l), /*offset=*/0, /*cpol=*/0);
}
#if __has_builtin(__builtin_amdgcn_s_wait_asynccnt)
#define WAIT_ASYNC(n) __builtin_amdgcn_s_wait_asynccnt(n)
#else
#define WAIT_ASYNC(n) asm volatile("s_wait_asynccnt %0" :: "i"(n) : "memory")
#endif
#else
#define HAVE_ASYNC_LDS 0
#endif

// D = A(16x32) * B(32x16) + C,  f16 in / f32 accumulate
__device__ __forceinline__ v8f wmma_f(v16h a, v16h b, v8f c) {
  return __builtin_amdgcn_wmma_f32_16x16x32_f16(
      /*neg_a=*/false, a, /*neg_b=*/false, b,
      /*c_mod=*/(short)0, c, /*reuse_a=*/false, /*reuse_b=*/false);
}

// A-fragment (16x32 f16, row-major, leading dim ld):
//   lanes 0-15 hold row M=lane, K = 0..7 and 16..23
//   lanes 16-31 hold row M=lane-16, K = 8..15 and 24..31   (ISA 7.12.2)
__device__ __forceinline__ v16h load_frag_a(const _Float16* tile, int ld, int lane) {
  int r = lane & 15, g = lane >> 4;
  const _Float16* p = tile + r * ld + g * 8;
  v16h f;
  *reinterpret_cast<v8h*>(&f)       = *reinterpret_cast<const v8h*>(p);
  *(reinterpret_cast<v8h*>(&f) + 1) = *reinterpret_cast<const v8h*>(p + 16);
  return f;
}

// B-fragment (32x16 f16). tileT: row j = column j of B, K contiguous (ld).
//   lanes 0-15 hold col N=lane, K=0..7 and 16..23 ; lanes 16-31 hold col N=lane-16, K=8..15 and 24..31
__device__ __forceinline__ v16h load_frag_b(const _Float16* tileT, int ld, int lane) {
  int j = lane & 15, g = lane >> 4;
  const _Float16* p = tileT + j * ld + g * 8;
  v16h f;
  *reinterpret_cast<v8h*>(&f)       = *reinterpret_cast<const v8h*>(p);
  *(reinterpret_cast<v8h*>(&f) + 1) = *reinterpret_cast<const v8h*>(p + 16);
  return f;
}

// ---------------------------------------------------------------------------
// K0: convert weights fp32 -> f16
// ---------------------------------------------------------------------------
__global__ void convw_kernel(const float* __restrict__ wq, const float* __restrict__ wp,
                             _Float16* __restrict__ wqh, _Float16* __restrict__ wph) {
  int i = blockIdx.x * 256 + threadIdx.x;           // grid covers C3*C exactly
  wqh[i] = (_Float16)wq[i];
  if (i < C * C) wph[i] = (_Float16)wp[i];
}

// ---------------------------------------------------------------------------
// K1: GroupNorm.  One block per (b, group): 8 channels x 4096 px.
// Writes xn fp32 [b][c][n] (residual) and xh f16 [b][n][c] (GEMM operand).
// ---------------------------------------------------------------------------
__global__ __launch_bounds__(256) void gn_kernel(
    const float* __restrict__ x, const float* __restrict__ scale,
    const float* __restrict__ bias, float* __restrict__ xn,
    _Float16* __restrict__ xh) {
  const int bg = blockIdx.x;                 // 0..B*GROUPS-1
  const int b = bg >> 5, g = bg & 31;
  const int CPG = C / GROUPS;                // 8
  const int M = CPG * N;                     // 32768
  const float* xp = x + ((size_t)b * C + (size_t)g * CPG) * N;

  float s = 0.f, ss = 0.f;
  for (int e = threadIdx.x; e < M; e += 256) {
    float v = xp[e];
    s += v; ss += v * v;
  }
  __shared__ float rs[256], rss[256];
  rs[threadIdx.x] = s; rss[threadIdx.x] = ss;
  __syncthreads();
  for (int w = 128; w > 0; w >>= 1) {
    if (threadIdx.x < w) {
      rs[threadIdx.x]  += rs[threadIdx.x + w];
      rss[threadIdx.x] += rss[threadIdx.x + w];
    }
    __syncthreads();
  }
  const float mean = rs[0] / (float)M;
  const float var  = rss[0] / (float)M - mean * mean;
  const float inv  = rsqrtf(var + EPS);

  for (int e = threadIdx.x; e < M; e += 256) {
    int cl = e >> 12, n = e & (N - 1);
    int c = g * CPG + cl;
    float v = (xp[e] - mean) * inv;
    v = v * scale[c] + bias[c];
    xn[((size_t)b * C + c) * N + n] = v;
    xh[((size_t)b * N + n) * C + c] = (_Float16)v;
  }
}

// ---------------------------------------------------------------------------
// K2: QKV projection.  qkvT[n][o] = sum_c xh[n][c] * w_qkv[o][c].
// One 16x16 tile per wave; K-loop = 8 WMMAs.
// Outputs: q,k as f16 [b][n][c];  v as f16 [b][c][n] (so the attention kernel
// can async-copy V tiles straight into transposed LDS layout).
// ---------------------------------------------------------------------------
__global__ __launch_bounds__(256) void qkv_kernel(
    const _Float16* __restrict__ xh, const _Float16* __restrict__ wq,
    _Float16* __restrict__ qb, _Float16* __restrict__ kb, _Float16* __restrict__ vb) {
  const int wave = threadIdx.x >> 5, lane = threadIdx.x & 31;
  const int tile = blockIdx.x * 8 + wave;
  const int TPB = (N / 16) * (C3 / 16);            // 12288 tiles / batch
  const int b = tile / TPB, t = tile % TPB;
  const int ni = (t & 255) * 16;                   // N/16 = 256
  const int oi = (t >> 8) * 16;

  const _Float16* A = xh + ((size_t)b * N + ni) * C;
  v8f acc = {};
#pragma unroll
  for (int kk = 0; kk < C; kk += 32) {
    v16h a  = load_frag_a(A + kk, C, lane);
    v16h bf = load_frag_b(wq + (size_t)oi * C + kk, C, lane);
    acc = wmma_f(a, bf, acc);
  }
  const int seg = oi >> 8;                          // 0=q 1=k 2=v (uniform)
  const int cc  = oi - seg * C + (lane & 15);
  const int hi8 = (lane >> 4) * 8;
  if (seg == 2) {                                   // V: channel-major [b][c][n]
#pragma unroll
    for (int p = 0; p < 8; ++p) {
      int row = ni + p + hi8;
      vb[((size_t)b * C + cc) * N + row] = (_Float16)acc[p];
    }
  } else {
    _Float16* dst = (seg == 0) ? qb : kb;
#pragma unroll
    for (int p = 0; p < 8; ++p) {
      int row = ni + p + hi8;
      dst[((size_t)b * N + row) * C + cc] = (_Float16)acc[p];
    }
  }
}

// ---------------------------------------------------------------------------
// K3: Flash attention.  Each wave owns 16 query rows; block (8 waves) shares
// double-buffered 32-key K/V LDS tiles filled by async global->LDS b128 DMA.
// Online softmax; O kept in 128 VGPRs f32.
// ---------------------------------------------------------------------------
#define KVH (32 * C + C * 32)   // halves per K+V buffer (16 KB + 16 KB)
#define JT  (N / 32)            // 128 key-tile iterations

#if HAVE_ASYNC_LDS
__device__ __forceinline__ void issue_tile(const _Float16* kb, const _Float16* vb,
                                           int b, int j0, int tid, _Float16* kl) {
  _Float16* vl = kl + 32 * C;
  const _Float16* ks = kb + ((size_t)b * N + j0) * C;    // 32 rows x 256 halves, contiguous
#pragma unroll
  for (int i = 0; i < 4; ++i) {
    int idx = (tid + i * 256) * 8;
    async_b128(ks + idx, kl + idx);
  }
  // V: row c = tid of [c][n] layout -> 32 contiguous halves into v_lds[c][0..31]
  const _Float16* vs = vb + ((size_t)b * C + tid) * N + j0;
#pragma unroll
  for (int i = 0; i < 4; ++i)
    async_b128(vs + i * 8, vl + tid * 32 + i * 8);
}
#endif

__global__ __launch_bounds__(256) void attn_kernel(
    const _Float16* __restrict__ qb, const _Float16* __restrict__ kb,
    const _Float16* __restrict__ vb, _Float16* __restrict__ ob) {
  extern __shared__ _Float16 smem[];
  // [buf0: K(32x256) V(256x32)] [buf1: same] [p_lds: 8 waves x 16 x 32]
  _Float16* p_lds = smem + 2 * KVH;

  const int tid = threadIdx.x, lane = tid & 31, wave = tid >> 5;
  const int g = lane >> 4, j16 = lane & 15;
  const int b = blockIdx.x >> 5, qblk = blockIdx.x & 31;
  const int qrow0 = qblk * 128 + wave * 16;

  // Cache this wave's 16x256 Q tile as 8 A-fragments (64 VGPRs/lane).
  const _Float16* qbase = qb + ((size_t)b * N + qrow0) * C;
  v16h qf[8];
#pragma unroll
  for (int kk = 0; kk < 8; ++kk) qf[kk] = load_frag_a(qbase + kk * 32, C, lane);

  float rm[8], rl[8];
#pragma unroll
  for (int p = 0; p < 8; ++p) { rm[p] = -3.0e38f; rl[p] = 0.f; }
  v8f oacc[16];
  const v8f vzero = {};
#pragma unroll
  for (int ct = 0; ct < 16; ++ct) oacc[ct] = vzero;

#if HAVE_ASYNC_LDS
  issue_tile(kb, vb, b, 0, tid, smem);               // prologue: tile 0 -> buf0
#endif

  for (int jt = 0; jt < JT; ++jt) {
    const int cur = jt & 1;
    _Float16* kl = smem + cur * KVH;
    _Float16* vl = kl + 32 * C;

#if HAVE_ASYNC_LDS
    if (jt + 1 < JT) {
      issue_tile(kb, vb, b, (jt + 1) * 32, tid, smem + (cur ^ 1) * KVH);
      WAIT_ASYNC(8);                                 // drain tile jt (in-order)
    } else {
      WAIT_ASYNC(0);
    }
    __syncthreads();                                 // tile jt visible to all waves
#else
    const int j0 = jt * 32;
    __syncthreads();
    {
      const v8h* ksrc = (const v8h*)(kb + ((size_t)b * N + j0) * C);
      v8h* kdst = (v8h*)kl;
      for (int idx = tid; idx < 32 * C / 8; idx += 256) kdst[idx] = ksrc[idx];
      const v8h* vsrc = (const v8h*)(vb + ((size_t)b * C + tid) * N + j0);
      v8h* vdst = (v8h*)(vl + tid * 32);
#pragma unroll
      for (int i = 0; i < 4; ++i) vdst[i] = vsrc[i];
      if (jt + 1 < JT) {
        __builtin_prefetch(kb + ((size_t)b * N + j0 + 32) * C + tid * 32, 0, 1);
        __builtin_prefetch(vb + ((size_t)b * C + tid) * N + j0 + 32, 0, 1);
      }
    }
    __syncthreads();
#endif

    // S = Q * K^T : two 16x16 score tiles (j lo / j hi), 16 WMMAs.
    v8f s0 = vzero, s1 = vzero;
#pragma unroll
    for (int kk = 0; kk < 8; ++kk) {
      v16h b0 = load_frag_b(kl + kk * 32, C, lane);
      v16h b1 = load_frag_b(kl + 16 * C + kk * 32, C, lane);
      s0 = wmma_f(qf[kk], b0, s0);
      s1 = wmma_f(qf[kk], b1, s1);
    }

    // Online softmax (row r lives in VGPR r across a 16-lane half-group).
    _Float16* pw = p_lds + wave * (16 * 32);
#pragma unroll
    for (int p = 0; p < 8; ++p) {
      float a0 = s0[p] * SCALE, a1 = s1[p] * SCALE;
      float mx = fmaxf(a0, a1);
      mx = fmaxf(mx, __shfl_xor(mx, 1, 32));
      mx = fmaxf(mx, __shfl_xor(mx, 2, 32));
      mx = fmaxf(mx, __shfl_xor(mx, 4, 32));
      mx = fmaxf(mx, __shfl_xor(mx, 8, 32));
      float mnew  = fmaxf(rm[p], mx);
      float alpha = __expf(rm[p] - mnew);
      float e0 = __expf(a0 - mnew), e1 = __expf(a1 - mnew);
      float rsum = e0 + e1;
      rsum += __shfl_xor(rsum, 1, 32);
      rsum += __shfl_xor(rsum, 2, 32);
      rsum += __shfl_xor(rsum, 4, 32);
      rsum += __shfl_xor(rsum, 8, 32);
      rl[p] = rl[p] * alpha + rsum;
      rm[p] = mnew;
#pragma unroll
      for (int ct = 0; ct < 16; ++ct) oacc[ct][p] *= alpha;
      int row = p + 8 * g;                             // D-layout row
      pw[row * 32 + j16]      = (_Float16)e0;
      pw[row * 32 + j16 + 16] = (_Float16)e1;
    }

    // O += P(16x32) * V(32x256): 16 WMMAs. P re-read in A-fragment layout.
    v16h pf = load_frag_a(pw, 32, lane);
#pragma unroll
    for (int ct = 0; ct < 16; ++ct) {
      v16h vf = load_frag_b(vl + ct * 16 * 32, 32, lane);
      oacc[ct] = wmma_f(pf, vf, oacc[ct]);
    }
    __syncthreads();                                   // buffer free for reuse
  }

  // Epilogue: O /= l, store f16 [b][n][c].
#pragma unroll
  for (int p = 0; p < 8; ++p) rl[p] = 1.0f / rl[p];
#pragma unroll
  for (int ct = 0; ct < 16; ++ct) {
#pragma unroll
    for (int p = 0; p < 8; ++p) {
      int row = qrow0 + p + 8 * g, col = ct * 16 + j16;
      ob[((size_t)b * N + row) * C + col] = (_Float16)(oacc[ct][p] * rl[p]);
    }
  }
}

// ---------------------------------------------------------------------------
// K4: output projection + bias + residual (writes final fp32 [b][c][h][w]).
// ---------------------------------------------------------------------------
__global__ __launch_bounds__(256) void proj_kernel(
    const _Float16* __restrict__ ob, const _Float16* __restrict__ wp,
    const float* __restrict__ bproj, const float* __restrict__ xn,
    float* __restrict__ out) {
  const int wave = threadIdx.x >> 5, lane = threadIdx.x & 31;
  const int tile = blockIdx.x * 8 + wave;
  const int TPB = (C / 16) * (N / 16);               // 4096 tiles / batch
  const int b = tile / TPB, t = tile % TPB;
  const int ni = (t & 255) * 16;
  const int ci = (t >> 8) * 16;

  const _Float16* Bb = ob + ((size_t)b * N + ni) * C;
  v8f acc = {};
#pragma unroll
  for (int kk = 0; kk < C; kk += 32) {
    v16h a  = load_frag_a(wp + (size_t)ci * C + kk, C, lane);
    v16h bf = load_frag_b(Bb + kk, C, lane);
    acc = wmma_f(a, bf, acc);
  }
  const int j16 = lane & 15, hi8 = (lane >> 4) * 8;
  const int n = ni + j16;
#pragma unroll
  for (int p = 0; p < 8; ++p) {
    int m = ci + p + hi8;
    size_t idx = ((size_t)b * C + m) * N + n;
    out[idx] = acc[p] + bproj[m] + xn[idx];
  }
}

// ---------------------------------------------------------------------------
// Launch
// ---------------------------------------------------------------------------
extern "C" void kernel_launch(void* const* d_in, const int* in_sizes, int n_in,
                              void* d_out, int out_size, void* d_ws, size_t ws_size,
                              hipStream_t stream) {
  const float* x     = (const float*)d_in[0];
  const float* nsc   = (const float*)d_in[1];
  const float* nbi   = (const float*)d_in[2];
  const float* wqkv  = (const float*)d_in[3];
  const float* wproj = (const float*)d_in[4];
  const float* bproj = (const float*)d_in[5];
  float* out = (float*)d_out;

  char* ws = (char*)d_ws;
  size_t off = 0;
  float*    xn    = (float*)(ws + off);    off += (size_t)B * C * N * sizeof(float);
  _Float16* xh    = (_Float16*)(ws + off); off += (size_t)B * N * C * sizeof(_Float16);
  _Float16* qb    = (_Float16*)(ws + off); off += (size_t)B * N * C * sizeof(_Float16);
  _Float16* kb    = (_Float16*)(ws + off); off += (size_t)B * N * C * sizeof(_Float16);
  _Float16* vb    = (_Float16*)(ws + off); off += (size_t)B * C * N * sizeof(_Float16);
  _Float16* obuf  = (_Float16*)(ws + off); off += (size_t)B * N * C * sizeof(_Float16);
  _Float16* wqh   = (_Float16*)(ws + off); off += (size_t)C3 * C * sizeof(_Float16);
  _Float16* wph   = (_Float16*)(ws + off); off += (size_t)C * C * sizeof(_Float16);

  convw_kernel<<<(C3 * C) / 256, 256, 0, stream>>>(wqkv, wproj, wqh, wph);
  gn_kernel<<<B * GROUPS, 256, 0, stream>>>(x, nsc, nbi, xn, xh);
  qkv_kernel<<<B * (N / 16) * (C3 / 16) / 8, 256, 0, stream>>>(xh, wqh, qb, kb, vb);
  const size_t attn_lds = (size_t)(2 * KVH + 8 * 16 * 32) * sizeof(_Float16); // 72 KB
  attn_kernel<<<B * (N / 128), 256, attn_lds, stream>>>(qb, kb, vb, obuf);
  proj_kernel<<<B * (C / 16) * (N / 16) / 8, 256, 0, stream>>>(obuf, wph, bproj, xn, out);
}
